// RBFKernelLayer_65481071400015
// MI455X (gfx1250) — matrix operand order repaired and checked by reading, not compiled, not run on hardware
//
#include <hip/hip_runtime.h>
#include <hip/hip_bf16.h>

// ---------------------------------------------------------------------------
// RBF kernel layer: out[n][m] = exp(-||x_n - c_m||^2)
//   = exp(-(||x||^2 + ||c||^2 - 2 x.c))
// Cross-term GEMM with V_WMMA_F32_16X16X32_BF16, bf16x3 split precision
// (x = hi + lo; acc += hi*hi + lo*hi + hi*lo in f32).
// Pre-pass converts f32 -> bf16 hi/lo once (in d_ws) so the GEMM inner loop
// is pure load/ds/wmma with no conversion VALU.
// ---------------------------------------------------------------------------

#define N_ROWS 16384
#define M_COLS 4096
#define DDIM   512

#define BM 128      // rows of x per workgroup
#define BN 64       // centroid columns per workgroup
#define BK 32       // K block (one bf16 WMMA depth)
#define LSTR 40     // LDS row stride in bf16 elems (80 B: 16B-aligned, 20-bank step)

typedef __attribute__((ext_vector_type(16))) __bf16 v16bf;
typedef __attribute__((ext_vector_type(8)))  float  v8f;

union FragBF {
    v16bf v;
    uint4 q[2];
};

// round-to-nearest-even f32 -> bf16 (hi) + bf16 of residual (lo)
__device__ __forceinline__ void split_bf16(float v, unsigned short& h, unsigned short& l) {
    unsigned u  = __float_as_uint(v);
    unsigned rh = u + 0x7FFFu + ((u >> 16) & 1u);
    h = (unsigned short)(rh >> 16);
    float hf = __uint_as_float(((unsigned)h) << 16);
    float r  = v - hf;
    unsigned u2 = __float_as_uint(r);
    unsigned rl = u2 + 0x7FFFu + ((u2 >> 16) & 1u);
    l = (unsigned short)(rl >> 16);
}

__device__ __forceinline__ void store_pair(unsigned short* hA, unsigned short* lA,
                                           int idx, float v0, float v1) {
    unsigned short h0, l0, h1, l1;
    split_bf16(v0, h0, l0);
    split_bf16(v1, h1, l1);
    *(unsigned int*)(hA + idx) = (unsigned)h0 | ((unsigned)h1 << 16);
    *(unsigned int*)(lA + idx) = (unsigned)l0 | ((unsigned)l1 << 16);
}

// A fragment (16x32 bf16, MxK): lane holds row (lane&15); K chunks of 8 at
// (lane>>4)*8 and (lane>>4)*8+16. Two 16B LDS reads.
__device__ __forceinline__ v16bf load_a_frag(const unsigned short* base, int lane) {
    int r  = lane & 15;
    int kh = (lane >> 4) << 3;              // 0 or 8
    const unsigned short* p = base + r * LSTR + kh;
    FragBF f;
    f.q[0] = *(const uint4*)(p);            // K = kh .. kh+7
    f.q[1] = *(const uint4*)(p + 16);       // K = kh+16 .. kh+23
    return f.v;
}

// B fragment (32x16 bf16, KxN): lane holds col (lane&15); K chunks of 8 at
// (lane>>4)*8 and (lane>>4)*8+16. Two 16B LDS reads.
__device__ __forceinline__ v16bf load_b_frag(const unsigned short* base, int lane) {
    int c  = lane & 15;
    int kb = (lane >> 4) << 3;              // 0 or 8
    const unsigned short* p = base + c * LSTR + kb;
    FragBF f;
    f.q[0] = *(const uint4*)(p);
    f.q[1] = *(const uint4*)(p + 16);
    return f.v;
}

__device__ __forceinline__ v8f wmma_bf16(v16bf a, v16bf b, v8f c) {
    return __builtin_amdgcn_wmma_f32_16x16x32_bf16(
        /*neg_a=*/false, a, /*neg_b=*/false, b,
        /*c_mod=*/(short)0, c, /*reuse_a=*/false, /*reuse_b=*/false);
}

// ---------------------------------------------------------------------------
// Pre-pass 1: squared L2 norms of rows (D = 512), one wave per row.
// ---------------------------------------------------------------------------
__global__ __launch_bounds__(256) void row_norms_kernel(const float* __restrict__ src,
                                                        float* __restrict__ dst, int rows) {
    const int lane = threadIdx.x & 31;
    const int wid  = threadIdx.x >> 5;
    const int row  = blockIdx.x * 8 + wid;
    if (row >= rows) return;
    const float4* p = (const float4*)(src + (size_t)row * DDIM);
    float s = 0.0f;
    #pragma unroll
    for (int i = 0; i < 4; ++i) {
        float4 v = p[lane + i * 32];
        s += v.x * v.x + v.y * v.y + v.z * v.z + v.w * v.w;
    }
    #pragma unroll
    for (int off = 16; off > 0; off >>= 1) s += __shfl_xor(s, off, 32);
    if (lane == 0) dst[row] = s;
}

// ---------------------------------------------------------------------------
// Pre-pass 2: f32 -> bf16 hi/lo split (8 elems / thread; 32B in, 16B+16B out)
// ---------------------------------------------------------------------------
__global__ __launch_bounds__(256) void split_kernel(const float* __restrict__ src,
                                                    unsigned short* __restrict__ hi,
                                                    unsigned short* __restrict__ lo,
                                                    int n8) {
    int i = blockIdx.x * 256 + threadIdx.x;
    if (i >= n8) return;
    const float4* p = (const float4*)(src) + (size_t)i * 2;
    float4 a = p[0], b = p[1];
    unsigned short hs[8], ls[8];
    split_bf16(a.x, hs[0], ls[0]); split_bf16(a.y, hs[1], ls[1]);
    split_bf16(a.z, hs[2], ls[2]); split_bf16(a.w, hs[3], ls[3]);
    split_bf16(b.x, hs[4], ls[4]); split_bf16(b.y, hs[5], ls[5]);
    split_bf16(b.z, hs[6], ls[6]); split_bf16(b.w, hs[7], ls[7]);
    uint4 hv, lv;
    hv.x = hs[0] | ((unsigned)hs[1] << 16); hv.y = hs[2] | ((unsigned)hs[3] << 16);
    hv.z = hs[4] | ((unsigned)hs[5] << 16); hv.w = hs[6] | ((unsigned)hs[7] << 16);
    lv.x = ls[0] | ((unsigned)ls[1] << 16); lv.y = ls[2] | ((unsigned)ls[3] << 16);
    lv.z = ls[4] | ((unsigned)ls[5] << 16); lv.w = ls[6] | ((unsigned)ls[7] << 16);
    ((uint4*)hi)[i] = hv;
    ((uint4*)lo)[i] = lv;
}

// ---------------------------------------------------------------------------
// Main fused kernel: bf16x3 WMMA GEMM + RBF epilogue.
// Grid: (N/BM, M/BN) = (128, 64); block: 256 threads = 8 waves (4 rows x 2 cols).
// PRE=true : A0/A1 = xh/xl, B0/B1 = ch/cl (bf16 in d_ws), pure copy staging.
// PRE=false: A0 = x f32, B0 = c f32; split to bf16 during staging (fallback).
// ---------------------------------------------------------------------------
template <bool PRE>
__global__ __launch_bounds__(256) void rbf_wmma_kernel(const void* __restrict__ A0,
                                                       const void* __restrict__ A1,
                                                       const void* __restrict__ B0,
                                                       const void* __restrict__ B1,
                                                       const float* __restrict__ xsq,
                                                       const float* __restrict__ csq,
                                                       float* __restrict__ out) {
    __shared__ unsigned short Ah[BM * LSTR];
    __shared__ unsigned short Al[BM * LSTR];
    __shared__ unsigned short Bh[BN * LSTR];
    __shared__ unsigned short Bl[BN * LSTR];

    const int tid  = threadIdx.x;
    const int lane = tid & 31;
    const int wid  = tid >> 5;
    const int wr   = wid >> 1;   // 0..3 -> 32-row strip
    const int wc   = wid & 1;    // 0..1 -> 32-col strip

    const int rowBase = blockIdx.x * BM;
    const int colBase = blockIdx.y * BN;

    v8f acc[2][2] = {};

    // ---- staging assignments ----
    // PRE path: A: 256 threads = (hi/lo sel = tid>>7) x 128 rows, 64 B each.
    //           B: (half = tid>>7) x (hi/lo sel = (tid>>6)&1) x 64 cols, 32 B each.
    const int p_asel = tid >> 7;                 // 0 = hi, 1 = lo  (uniform per wave)
    const int p_arow = tid & 127;
    const int p_bsel = (tid >> 6) & 1;           // uniform per wave
    const int p_bcol = tid & 63;
    const int p_bhalf = tid >> 7;                // 0/1 -> k-half (uniform per wave)

    const unsigned short* pa_src = PRE
        ? (const unsigned short*)(p_asel ? A1 : A0) + (size_t)(rowBase + p_arow) * DDIM
        : nullptr;
    unsigned short* pa_dst = (p_asel ? Al : Ah) + p_arow * LSTR;
    const unsigned short* pb_src = PRE
        ? (const unsigned short*)(p_bsel ? B1 : B0) + (size_t)(colBase + p_bcol) * DDIM + p_bhalf * 16
        : nullptr;
    unsigned short* pb_dst = (p_bsel ? Bl : Bh) + p_bcol * LSTR + p_bhalf * 16;

    // Fallback (f32-convert) path assignments
    const int f_ar  = tid >> 1;          // A row (0..127), 2 threads/row, 16 floats
    const int f_aks = (tid & 1) * 16;
    const int f_bc  = tid >> 2;          // B col (0..63), 4 threads/col, 8 floats
    const int f_bks = (tid & 3) * 8;
    const float* f_xp = PRE ? nullptr
        : (const float*)A0 + (size_t)(rowBase + f_ar) * DDIM + f_aks;
    const float* f_cp = PRE ? nullptr
        : (const float*)B0 + (size_t)(colBase + f_bc) * DDIM + f_bks;

    for (int k0 = 0; k0 < DDIM; k0 += BK) {
        if (PRE) {
            uint4 abuf[4], bbuf[2];
            #pragma unroll
            for (int q = 0; q < 4; ++q) abuf[q] = *(const uint4*)(pa_src + k0 + q * 8);
            #pragma unroll
            for (int q = 0; q < 2; ++q) bbuf[q] = *(const uint4*)(pb_src + k0 + q * 8);
            __syncthreads();   // previous iteration done reading LDS
            #pragma unroll
            for (int q = 0; q < 4; ++q) *(uint4*)(pa_dst + q * 8) = abuf[q];
            #pragma unroll
            for (int q = 0; q < 2; ++q) *(uint4*)(pb_dst + q * 8) = bbuf[q];
        } else {
            float4 areg[4], breg[2];
            #pragma unroll
            for (int q = 0; q < 4; ++q) areg[q] = *(const float4*)(f_xp + k0 + q * 4);
            #pragma unroll
            for (int q = 0; q < 2; ++q) breg[q] = *(const float4*)(f_cp + k0 + q * 4);
            __syncthreads();
            const int abase = f_ar * LSTR + f_aks;
            #pragma unroll
            for (int q = 0; q < 4; ++q) {
                store_pair(Ah, Al, abase + q * 4 + 0, areg[q].x, areg[q].y);
                store_pair(Ah, Al, abase + q * 4 + 2, areg[q].z, areg[q].w);
            }
            const int bbase = f_bc * LSTR + f_bks;
            #pragma unroll
            for (int q = 0; q < 2; ++q) {
                store_pair(Bh, Bl, bbase + q * 4 + 0, breg[q].x, breg[q].y);
                store_pair(Bh, Bl, bbase + q * 4 + 2, breg[q].z, breg[q].w);
            }
        }

        __syncthreads();

        // ---- fragments + 12 WMMAs (bf16x3 over 2x2 tiles) ----
        v16bf fah[2], fal[2], fbh[2], fbl[2];
        #pragma unroll
        for (int mt = 0; mt < 2; ++mt) {
            const int ro = (wr * 32 + mt * 16) * LSTR;
            fah[mt] = load_a_frag(Ah + ro, lane);
            fal[mt] = load_a_frag(Al + ro, lane);
        }
        #pragma unroll
        for (int nt = 0; nt < 2; ++nt) {
            const int co = (wc * 32 + nt * 16) * LSTR;
            fbh[nt] = load_b_frag(Bh + co, lane);
            fbl[nt] = load_b_frag(Bl + co, lane);
        }
        #pragma unroll
        for (int mt = 0; mt < 2; ++mt) {
            #pragma unroll
            for (int nt = 0; nt < 2; ++nt) {
                acc[mt][nt] = wmma_bf16(fah[mt], fbh[nt], acc[mt][nt]);
                acc[mt][nt] = wmma_bf16(fal[mt], fbh[nt], acc[mt][nt]);
                acc[mt][nt] = wmma_bf16(fah[mt], fbl[nt], acc[mt][nt]);
            }
        }
    }

    // ---- epilogue: dist^2 = ||x||^2 + ||c||^2 - 2*cross ; out = exp(-dist^2)
    #pragma unroll
    for (int mt = 0; mt < 2; ++mt) {
        #pragma unroll
        for (int nt = 0; nt < 2; ++nt) {
            const int col = colBase + wc * 32 + nt * 16 + (lane & 15);
            const float cs = csq[col];
            #pragma unroll
            for (int j = 0; j < 8; ++j) {
                const int row = rowBase + wr * 32 + mt * 16 + j + ((lane >> 4) << 3);
                float d = xsq[row] + cs - 2.0f * acc[mt][nt][j];
                d = fmaxf(d, 0.0f);
                out[(size_t)row * M_COLS + col] = __expf(-d);   // GAMMA = 1.0
            }
        }
    }
}

// ---------------------------------------------------------------------------
extern "C" void kernel_launch(void* const* d_in, const int* in_sizes, int n_in,
                              void* d_out, int out_size, void* d_ws, size_t ws_size,
                              hipStream_t stream) {
    const float* x = (const float*)d_in[0];   // [16384, 512] f32
    const float* c = (const float*)d_in[1];   // [4096, 512]  f32
    float* out = (float*)d_out;               // [16384, 4096] f32

    float* xsq = (float*)d_ws;                // 16384 f32
    float* csq = xsq + N_ROWS;                // 4096 f32
    const size_t norm_bytes = (size_t)(N_ROWS + M_COLS) * sizeof(float);   // 80 KB
    const size_t xcnt = (size_t)N_ROWS * DDIM;                             // 8.4M
    const size_t ccnt = (size_t)M_COLS * DDIM;                             // 2.1M
    const size_t conv_bytes = (xcnt + ccnt) * 2 * sizeof(unsigned short);  // ~40 MB

    row_norms_kernel<<<N_ROWS / 8, 256, 0, stream>>>(x, xsq, N_ROWS);
    row_norms_kernel<<<M_COLS / 8, 256, 0, stream>>>(c, csq, M_COLS);

    dim3 grid(N_ROWS / BM, M_COLS / BN);

    if (ws_size >= norm_bytes + conv_bytes) {
        unsigned short* xh = (unsigned short*)((char*)d_ws + norm_bytes);
        unsigned short* xl = xh + xcnt;
        unsigned short* ch = xl + xcnt;
        unsigned short* cl = ch + ccnt;
        split_kernel<<<(int)(xcnt / 8 / 256), 256, 0, stream>>>(x, xh, xl, (int)(xcnt / 8));
        split_kernel<<<(int)(ccnt / 8 / 256), 256, 0, stream>>>(c, ch, cl, (int)(ccnt / 8));
        rbf_wmma_kernel<true><<<grid, 256, 0, stream>>>(xh, xl, ch, cl, xsq, csq, out);
    } else {
        rbf_wmma_kernel<false><<<grid, 256, 0, stream>>>(x, nullptr, c, nullptr, xsq, csq, out);
    }
}
